// WinCoTransLayer_249108103597
// MI455X (gfx1250) — matrix operand order repaired and checked by reading, not compiled, not run on hardware
//
#include <hip/hip_runtime.h>
#include <hip/hip_bf16.h>
#include <math.h>

typedef __attribute__((ext_vector_type(8)))  float        v8f;
typedef __attribute__((ext_vector_type(16))) __bf16       v16bf;
typedef __attribute__((ext_vector_type(8)))  __bf16       v8bf;
typedef __attribute__((ext_vector_type(4)))  unsigned int u32x4;
typedef __attribute__((ext_vector_type(8)))  int          i32x8;
typedef __attribute__((ext_vector_type(4)))  int          i32x4;

// Problem constants
constexpr int CB   = 2;            // batch
constexpr int CH   = 128;          // H
constexpr int CW   = 128;          // W
constexpr int CC   = 512;          // C
constexpr int CN   = 64;           // tokens per window
constexpr int CNWIN = CB * 16 * 16;      // 512 windows
constexpr int CM   = CB * CH * CW;       // 32768 tokens
constexpr int CHID = 2048;

constexpr int EPI_BF16 = 0, EPI_GELU = 1, EPI_PROJ = 2, EPI_RES = 3;

#if defined(__AMDGCN__) && __has_builtin(__builtin_amdgcn_tensor_load_to_lds) && __has_builtin(__builtin_amdgcn_s_wait_tensorcnt)
#define USE_TDM 1
#else
#define USE_TDM 0
#endif

// twiddles: TWC[i] = cos(2*pi*i/8), TWS[i] = -sin(2*pi*i/8)
__constant__ float TWC[8] = { 1.f,  0.70710678f, 0.f, -0.70710678f, -1.f, -0.70710678f, 0.f,  0.70710678f };
__constant__ float TWS[8] = { 0.f, -0.70710678f, -1.f, -0.70710678f, 0.f,  0.70710678f, 1.f,  0.70710678f };

__device__ __forceinline__ float gelu_f(float x) {
  return 0.5f * x * (1.0f + erff(x * 0.70710678f));
}

// window-order token -> original token index
__device__ __forceinline__ int win2orig(int wr) {
  int w = wr >> 6, t = wr & 63;
  int b = w >> 8, rem = w & 255;
  int wh = rem >> 4, wwi = rem & 15;
  int r = t >> 3, c = t & 7;
  return (b << 14) + ((wh * 8 + r) << 7) + wwi * 8 + c;
}
// original token -> window-order token index
__device__ __forceinline__ int orig2win(int t) {
  int b = t >> 14, l = t & 16383;
  int hh = l >> 7, wwi = l & 127;
  int w = (b << 8) + ((hh >> 3) << 4) + (wwi >> 3);
  return (w << 6) + ((hh & 7) << 3) + (wwi & 7);
}

// ---- WMMA fragment loaders (layouts per CDNA5 ISA 7.12.2) ----
// A: 16 (M) x 32 (K) bf16. row = lane&15; lanes<16: K[0..7],K[16..23]; lanes>=16: K[8..15],K[24..31]
__device__ __forceinline__ v16bf load_frag_a(const __bf16* A, int ld, int lane) {
  int r = lane & 15, half = lane >> 4;
  const __bf16* p = A + r * ld + half * 8;
  v8bf lo = *(const v8bf*)(p);
  v8bf hi = *(const v8bf*)(p + 16);
  return __builtin_shufflevector(lo, hi, 0,1,2,3,4,5,6,7,8,9,10,11,12,13,14,15);
}
// B: 32 (K) x 16 (N), stored as Bt[N][K] row-major. col = lane&15; lanes<16: K[0..7],K[16..23]; lanes>=16: K[8..15],K[24..31]
__device__ __forceinline__ v16bf load_frag_b(const __bf16* Bt, int ldk, int lane) {
  int c = lane & 15, half = lane >> 4;
  const __bf16* p = Bt + c * ldk + half * 8;
  v8bf lo = *(const v8bf*)(p);
  v8bf hi = *(const v8bf*)(p + 16);
  return __builtin_shufflevector(lo, hi, 0,1,2,3,4,5,6,7,8,9,10,11,12,13,14,15);
}

#if USE_TDM
// Issue a 2D TDM load: tile_h rows x tile_w bf16 elements, row stride = stride_elems,
// from global `gaddr` into LDS byte offset `lds_off`.  D# layout per CDNA5 ISA 10.8.
__device__ __forceinline__ void tdm_load_2d_bf16(unsigned lds_off, const void* gaddr,
                                                 unsigned tile_w, unsigned tile_h,
                                                 unsigned stride_elems) {
  unsigned long long ga = (unsigned long long)gaddr;
  u32x4 g0;
  g0[0] = 1u;                                       // count=1, user mode, no gather
  g0[1] = lds_off;                                  // lds_addr (bytes)
  g0[2] = (unsigned)(ga & 0xffffffffu);             // global_addr[31:0]
  g0[3] = (unsigned)((ga >> 32) & 0x01ffffffu) | (2u << 30);  // addr[56:32] | type=2
  unsigned dim0 = tile_w;                           // tensor bounds == tile (no OOB)
  unsigned dim1 = tile_h;
  i32x8 g1;
  g1[0] = (int)(1u << 16);                          // workgroup_mask=0, data_size=1 (2B)
  g1[1] = (int)((dim0 & 0xffffu) << 16);            // atomic_barrier=0 | tensor_dim0 lo16
  g1[2] = (int)((dim0 >> 16) | ((dim1 & 0xffffu) << 16));   // dim0 hi16 | dim1 lo16
  g1[3] = (int)((dim1 >> 16) | (tile_w << 16));     // dim1 hi16 | tile_dim0
  g1[4] = (int)(tile_h & 0xffffu);                  // tile_dim1 | tile_dim2=0
  g1[5] = (int)stride_elems;                        // tensor_dim0_stride[31:0]
  g1[6] = 0;                                        // stride hi | dim1_stride lo (unused, 2D)
  g1[7] = 0;
  i32x4 z4 = {0, 0, 0, 0};
#if __clang_major__ >= 23
  i32x8 z8 = {0, 0, 0, 0, 0, 0, 0, 0};
  __builtin_amdgcn_tensor_load_to_lds(g0, g1, z4, z4, z8, 0);
#else
  __builtin_amdgcn_tensor_load_to_lds(g0, g1, z4, z4, 0);
#endif
}
#endif

// ---------------- weight transpose + bf16 convert: W[K][N] -> Wt[N][K] ----------------
__global__ __launch_bounds__(256) void wconv_kernel(const float* __restrict__ W,
                                                    __bf16* __restrict__ Wt, int K, int N) {
  int i = blockIdx.x * 256 + threadIdx.x;
  if (i < K * N) {
    int n = i / K, k = i - n * K;
    Wt[i] = (__bf16)W[k * N + n];
  }
}

// ---------------- LayerNorm (wave per token), optional remap to window order ----------------
template <bool REMAP>
__global__ __launch_bounds__(256) void layernorm_kernel(const float* __restrict__ x,
                                                        const float* __restrict__ g,
                                                        const float* __restrict__ bb,
                                                        __bf16* __restrict__ out) {
  int tok  = (blockIdx.x * 256 + threadIdx.x) >> 5;
  int lane = threadIdx.x & 31;
  const float* xp = x + (size_t)tok * CC;
  float4 v[4];
  float s = 0.f;
#pragma unroll
  for (int i = 0; i < 4; ++i) {
    v[i] = ((const float4*)xp)[lane + i * 32];
    s += v[i].x + v[i].y + v[i].z + v[i].w;
  }
#pragma unroll
  for (int m = 16; m; m >>= 1) s += __shfl_xor(s, m, 32);
  float mu = s * (1.f / 512.f);
  float vs = 0.f;
#pragma unroll
  for (int i = 0; i < 4; ++i) {
    float dx = v[i].x - mu, dy = v[i].y - mu, dz = v[i].z - mu, dw = v[i].w - mu;
    vs += dx * dx + dy * dy + dz * dz + dw * dw;
  }
#pragma unroll
  for (int m = 16; m; m >>= 1) vs += __shfl_xor(vs, m, 32);
  float inv = rsqrtf(vs * (1.f / 512.f) + 1e-5f);
  int otok = REMAP ? orig2win(tok) : tok;
  __bf16* op = out + (size_t)otok * CC;
#pragma unroll
  for (int i = 0; i < 4; ++i) {
    int ch = (lane + i * 32) * 4;
    float a0 = (v[i].x - mu) * inv * g[ch + 0] + bb[ch + 0];
    float a1 = (v[i].y - mu) * inv * g[ch + 1] + bb[ch + 1];
    float a2 = (v[i].z - mu) * inv * g[ch + 2] + bb[ch + 2];
    float a3 = (v[i].w - mu) * inv * g[ch + 3] + bb[ch + 3];
    op[ch + 0] = (__bf16)a0; op[ch + 1] = (__bf16)a1;
    op[ch + 2] = (__bf16)a2; op[ch + 3] = (__bf16)a3;
  }
}

// ---------------- FFT conditioning: akv = irfft2(rfft2(win) * cond), ortho ----------------
__global__ __launch_bounds__(256) void fftcond_kernel(const __bf16* __restrict__ h,
                                                      const float* __restrict__ cond,
                                                      __bf16* __restrict__ akv) {
  int gid = blockIdx.x * 256 + threadIdx.x;
  int c = gid & (CC - 1), w = gid >> 9;
  float Rr[8][5], Ri[8][5];
  // row-wise rfft (width axis)
  for (int r = 0; r < 8; ++r) {
    float xr[8];
#pragma unroll
    for (int cc = 0; cc < 8; ++cc)
      xr[cc] = (float)h[((size_t)(w * CN + r * 8 + cc)) * CC + c];
#pragma unroll
    for (int k = 0; k < 5; ++k) {
      float sr = 0.f, si = 0.f;
#pragma unroll
      for (int cc = 0; cc < 8; ++cc) {
        int idx = (k * cc) & 7;
        sr += xr[cc] * TWC[idx];
        si += xr[cc] * TWS[idx];
      }
      Rr[r][k] = sr; Ri[r][k] = si;
    }
  }
  // column-wise full DFT (in place per k), then multiply cond
  for (int k = 0; k < 5; ++k) {
    float tr[8], ti[8];
#pragma unroll
    for (int u = 0; u < 8; ++u) {
      float sr = 0.f, si = 0.f;
#pragma unroll
      for (int r = 0; r < 8; ++r) {
        int idx = (u * r) & 7;
        float cr = TWC[idx], ci = TWS[idx];
        sr += Rr[r][k] * cr - Ri[r][k] * ci;
        si += Rr[r][k] * ci + Ri[r][k] * cr;
      }
      tr[u] = sr; ti[u] = si;
    }
#pragma unroll
    for (int u = 0; u < 8; ++u) {
      const float* cp = cond + (((size_t)(u * 5 + k)) * CC + c) * 2;
      float cr = cp[0], ci = cp[1];
      Rr[u][k] = tr[u] * cr - ti[u] * ci;
      Ri[u][k] = tr[u] * ci + ti[u] * cr;
    }
  }
  // inverse 2D (Hermitian half-spectrum), combined ortho scale 1/64
  for (int r = 0; r < 8; ++r)
    for (int cc = 0; cc < 8; ++cc) {
      float s = 0.f;
#pragma unroll
      for (int u = 0; u < 8; ++u)
#pragma unroll
        for (int k = 0; k < 5; ++k) {
          int idx = (u * r + k * cc) & 7;
          float t = Rr[u][k] * TWC[idx] + Ri[u][k] * TWS[idx];
          s += (k == 0 || k == 4) ? t : 2.f * t;
        }
      akv[((size_t)(w * CN + r * 8 + cc)) * CC + c] = (__bf16)(s * (1.f / 64.f));
    }
}

// ------- generic bf16 WMMA GEMM: out = A[M,K] * Wt[N,K]^T + bias, epilogue -------
// A tiles (64 x 32 bf16) are streamed HBM->LDS by the Tensor Data Mover (double
// buffered, TENSORcnt + workgroup barrier), B fragments come from L2 directly.
template <int EPI>
__global__ __launch_bounds__(256) void gemm_bf16_kernel(const __bf16* __restrict__ A,
                                                        const __bf16* __restrict__ Wt,
                                                        const float* __restrict__ bias,
                                                        void* __restrict__ out,
                                                        const float* __restrict__ res,
                                                        int N, int K) {
  int lane = threadIdx.x & 31;
  int wave = threadIdx.x >> 5;
  int mt = wave & 3, nh = wave >> 2;
  int mbase  = blockIdx.x * 64 + mt * 16;
  int nbase0 = blockIdx.y * 128 + nh * 64;
  v8f acc[4] = {};

#if USE_TDM
  __shared__ __attribute__((aligned(64))) __bf16 atile[2][64][32];
  const __bf16* Ablk = A + (size_t)(blockIdx.x * 64) * K;
  bool issuer = (wave == 0);
  unsigned lds_base = (unsigned)(size_t)(&atile[0][0][0]);
  if (issuer) {
    tdm_load_2d_bf16(lds_base, Ablk, 32, 64, (unsigned)K);
    __builtin_amdgcn_s_wait_tensorcnt(0);
  }
  __syncthreads();
  for (int kb = 0; kb < K; kb += 32) {
    int cur = (kb >> 5) & 1;
    if (issuer && (kb + 32 < K))
      tdm_load_2d_bf16(lds_base + (cur ^ 1) * 64 * 32 * 2, Ablk + kb + 32, 32, 64, (unsigned)K);
    v16bf af = load_frag_a(&atile[cur][mt * 16][0], 32, lane);
#pragma unroll
    for (int j = 0; j < 4; ++j) {
      v16bf bf = load_frag_b(Wt + (size_t)(nbase0 + j * 16) * K + kb, K, lane);
      acc[j] = __builtin_amdgcn_wmma_f32_16x16x32_bf16(false, af, false, bf,
                                                       (short)0, acc[j], false, false);
    }
    if (issuer) __builtin_amdgcn_s_wait_tensorcnt(0);
    __syncthreads();
  }
#else
  const __bf16* Arow = A + (size_t)mbase * K;
  for (int kb = 0; kb < K; kb += 32) {
    __builtin_prefetch(Arow + kb + 128, 0, 1);
    v16bf af = load_frag_a(Arow + kb, K, lane);
#pragma unroll
    for (int j = 0; j < 4; ++j) {
      v16bf bf = load_frag_b(Wt + (size_t)(nbase0 + j * 16) * K + kb, K, lane);
      acc[j] = __builtin_amdgcn_wmma_f32_16x16x32_bf16(false, af, false, bf,
                                                       (short)0, acc[j], false, false);
    }
  }
#endif

  int colo = lane & 15, rh = (lane >> 4) * 8;
#pragma unroll
  for (int j = 0; j < 4; ++j) {
    int col = nbase0 + j * 16 + colo;
    float bv = bias[col];
#pragma unroll
    for (int i = 0; i < 8; ++i) {
      int row = mbase + rh + i;
      float v = acc[j][i] + bv;
      if (EPI == EPI_BF16) {
        ((__bf16*)out)[(size_t)row * N + col] = (__bf16)v;
      } else if (EPI == EPI_GELU) {
        ((__bf16*)out)[(size_t)row * N + col] = (__bf16)gelu_f(v);
      } else if (EPI == EPI_PROJ) {
        int orow = win2orig(row);
        ((float*)out)[(size_t)orow * N + col] = v + res[(size_t)orow * N + col];
      } else { // EPI_RES
        ((float*)out)[(size_t)row * N + col] = v + res[(size_t)row * N + col];
      }
    }
  }
}

// ---------------- windowed attention: one block per (window, head) ----------------
__global__ __launch_bounds__(128) void attn_kernel(const __bf16* __restrict__ qb,
                                                   const __bf16* __restrict__ kvb,
                                                   const float* __restrict__ pos,
                                                   __bf16* __restrict__ ob) {
  __shared__ __attribute__((aligned(32))) __bf16 qs[64][64];
  __shared__ __attribute__((aligned(32))) __bf16 ks[64][64];
  __shared__ __attribute__((aligned(32))) __bf16 vt[64][64];   // transposed v: [chan][tok]
  __shared__ __attribute__((aligned(32))) float  Sm[64][64];
  __shared__ __attribute__((aligned(32))) __bf16 Pm[64][64];

  int w  = blockIdx.x >> 3;
  int hd = blockIdx.x & 7;
  int tid = threadIdx.x;
  int lane = tid & 31;
  int mt = tid >> 5;  // wave id 0..3

  // cooperative loads: q, k rows; v transposed
  for (int i = tid; i < 64 * 8; i += 128) {
    int t = i >> 3, c0 = (i & 7) * 8;
    *(v8bf*)&qs[t][c0] = *(const v8bf*)&qb[((size_t)(w * CN + t)) * CC + hd * 64 + c0];
    *(v8bf*)&ks[t][c0] = *(const v8bf*)&kvb[((size_t)(w * CN + t)) * (2 * CC) + CC + hd * 64 + c0];
    v8bf vv = *(const v8bf*)&kvb[((size_t)(w * CN + t)) * (2 * CC) + hd * 64 + c0];
#pragma unroll
    for (int j = 0; j < 8; ++j) vt[c0 + j][t] = vv[j];
  }
  __syncthreads();

  // S = (q * scale) @ k^T
  {
    v8f acc[4] = {};
#pragma unroll
    for (int kb = 0; kb < 64; kb += 32) {
      v16bf af = load_frag_a(&qs[mt * 16][0] + kb, 64, lane);
#pragma unroll
      for (int nt = 0; nt < 4; ++nt) {
        v16bf bf = load_frag_b(&ks[nt * 16][0] + kb, 64, lane);
        acc[nt] = __builtin_amdgcn_wmma_f32_16x16x32_bf16(false, af, false, bf,
                                                          (short)0, acc[nt], false, false);
      }
    }
    int colo = lane & 15, rh = (lane >> 4) * 8;
#pragma unroll
    for (int nt = 0; nt < 4; ++nt)
#pragma unroll
      for (int i = 0; i < 8; ++i)
        Sm[mt * 16 + rh + i][nt * 16 + colo] = acc[nt][i] * 0.125f;
  }
  __syncthreads();

  // row softmax with pos bias -> P (bf16)
  if (tid < 64) {
    int r = tid;
    float mx = -3.4e38f;
    float buf[64];
#pragma unroll 8
    for (int n = 0; n < 64; ++n) {
      float v = Sm[r][n] + pos[r * 64 + n];
      buf[n] = v;
      mx = fmaxf(mx, v);
    }
    float sum = 0.f;
#pragma unroll 8
    for (int n = 0; n < 64; ++n) { buf[n] = __expf(buf[n] - mx); sum += buf[n]; }
    float rs = 1.f / sum;
#pragma unroll 8
    for (int n = 0; n < 64; ++n) Pm[r][n] = (__bf16)(buf[n] * rs);
  }
  __syncthreads();

  // O = P @ v
  {
    v8f acc[4] = {};
#pragma unroll
    for (int kb = 0; kb < 64; kb += 32) {
      v16bf af = load_frag_a(&Pm[mt * 16][0] + kb, 64, lane);
#pragma unroll
      for (int nt = 0; nt < 4; ++nt) {
        v16bf bf = load_frag_b(&vt[nt * 16][0] + kb, 64, lane);
        acc[nt] = __builtin_amdgcn_wmma_f32_16x16x32_bf16(false, af, false, bf,
                                                          (short)0, acc[nt], false, false);
      }
    }
    int colo = lane & 15, rh = (lane >> 4) * 8;
#pragma unroll
    for (int nt = 0; nt < 4; ++nt)
#pragma unroll
      for (int i = 0; i < 8; ++i)
        ob[((size_t)(w * CN + mt * 16 + rh + i)) * CC + hd * 64 + nt * 16 + colo] =
            (__bf16)acc[nt][i];
  }
}

// ---------------- depthwise 3x3 conv + GELU over (B,H,W,HID) bf16 ----------------
__global__ __launch_bounds__(256) void dwconv_kernel(const __bf16* __restrict__ f,
                                                     const float* __restrict__ wgt,
                                                     const float* __restrict__ bias,
                                                     __bf16* __restrict__ g) {
  int gid = blockIdx.x * 256 + threadIdx.x;
  int cg  = gid & 255;           // 2048/8 channel groups
  int pix = gid >> 8;            // 0..32767
  int c0 = cg * 8;
  int b  = pix >> 14;
  int hh = (pix >> 7) & 127;
  int wx = pix & 127;
  float acc[8] = {0.f, 0.f, 0.f, 0.f, 0.f, 0.f, 0.f, 0.f};
#pragma unroll
  for (int dy = -1; dy <= 1; ++dy) {
    int y = hh + dy;
    if ((unsigned)y >= 128u) continue;
#pragma unroll
    for (int dx = -1; dx <= 1; ++dx) {
      int x = wx + dx;
      if ((unsigned)x >= 128u) continue;
      v8bf t = *(const v8bf*)&f[((size_t)((b << 14) + y * 128 + x)) * CHID + c0];
#pragma unroll
      for (int j = 0; j < 8; ++j)
        acc[j] += (float)t[j] * wgt[(c0 + j) * 9 + (dy + 1) * 3 + (dx + 1)];
    }
  }
  __bf16* gp = g + (size_t)pix * CHID + c0;
#pragma unroll
  for (int j = 0; j < 8; ++j)
    gp[j] = (__bf16)gelu_f(acc[j] + bias[c0 + j]);
}

// ================================================================================
extern "C" void kernel_launch(void* const* d_in, const int* in_sizes, int n_in,
                              void* d_out, int out_size, void* d_ws, size_t ws_size,
                              hipStream_t stream) {
  const float* x      = (const float*)d_in[0];
  const float* gamma1 = (const float*)d_in[1];
  const float* beta1  = (const float*)d_in[2];
  const float* Wq     = (const float*)d_in[3];
  const float* bq     = (const float*)d_in[4];
  const float* Wkv    = (const float*)d_in[5];
  const float* bkv    = (const float*)d_in[6];
  const float* cond   = (const float*)d_in[7];
  const float* pos    = (const float*)d_in[8];
  const float* Wp     = (const float*)d_in[9];
  const float* bp     = (const float*)d_in[10];
  const float* gamma2 = (const float*)d_in[11];
  const float* beta2  = (const float*)d_in[12];
  const float* W1     = (const float*)d_in[13];
  const float* b1     = (const float*)d_in[14];
  const float* dw_w   = (const float*)d_in[15];
  const float* dw_b   = (const float*)d_in[16];
  const float* W2     = (const float*)d_in[17];
  const float* b2     = (const float*)d_in[18];

  char* ws = (char*)d_ws;
  const size_t MB = 1ull << 20;
  __bf16* wqT  = (__bf16*)(ws + 0 * MB);     // 512 KB
  __bf16* wkvT = (__bf16*)(ws + 1 * MB);     // 1 MB
  __bf16* wpT  = (__bf16*)(ws + 2 * MB);     // 512 KB
  __bf16* w1T  = (__bf16*)(ws + 3 * MB);     // 2 MB
  __bf16* w2T  = (__bf16*)(ws + 5 * MB);     // 2 MB
  __bf16* hbuf = (__bf16*)(ws + 8 * MB);     // 32 MB  (h, later h2)
  __bf16* akv  = (__bf16*)(ws + 40 * MB);    // 32 MB  (akv, later attn_out; later part of g)
  __bf16* qbuf = (__bf16*)(ws + 72 * MB);    // 32 MB
  __bf16* kvb  = (__bf16*)(ws + 104 * MB);   // 64 MB
  float*  xmid = (float*)(ws + 168 * MB);    // 64 MB
  __bf16* fbuf = (__bf16*)(ws + 232 * MB);   // 128 MB
  __bf16* gbuf = (__bf16*)(ws + 40 * MB);    // 128 MB (overlays dead akv/q/kv)
  __bf16* attn_out = akv;

  // 1) weight transpose+convert to bf16 [N][K]
  wconv_kernel<<<dim3((CC * CC + 255) / 256), 256, 0, stream>>>(Wq, wqT, CC, CC);
  wconv_kernel<<<dim3((CC * 2 * CC + 255) / 256), 256, 0, stream>>>(Wkv, wkvT, CC, 2 * CC);
  wconv_kernel<<<dim3((CC * CC + 255) / 256), 256, 0, stream>>>(Wp, wpT, CC, CC);
  wconv_kernel<<<dim3((CC * CHID + 255) / 256), 256, 0, stream>>>(W1, w1T, CC, CHID);
  wconv_kernel<<<dim3((CHID * CC + 255) / 256), 256, 0, stream>>>(W2, w2T, CHID, CC);

  // 2) LN1 -> h (window order, bf16)
  layernorm_kernel<true><<<dim3(CM / 8), 256, 0, stream>>>(x, gamma1, beta1, hbuf);

  // 3) FFT conditioning -> akv
  fftcond_kernel<<<dim3(CNWIN * CC / 256), 256, 0, stream>>>(hbuf, cond, akv);

  // 4) Q = akv @ WqT + bq ; KV = h @ WkvT + bkv
  gemm_bf16_kernel<EPI_BF16><<<dim3(CM / 64, CC / 128), 256, 0, stream>>>(
      akv, wqT, bq, (void*)qbuf, nullptr, CC, CC);
  gemm_bf16_kernel<EPI_BF16><<<dim3(CM / 64, 2 * CC / 128), 256, 0, stream>>>(
      hbuf, wkvT, bkv, (void*)kvb, nullptr, 2 * CC, CC);

  // 5) windowed attention
  attn_kernel<<<dim3(CNWIN * 8), 128, 0, stream>>>(qbuf, kvb, pos, attn_out);

  // 6) proj + window-reverse + residual -> xmid (fp32)
  gemm_bf16_kernel<EPI_PROJ><<<dim3(CM / 64, CC / 128), 256, 0, stream>>>(
      attn_out, wpT, bp, (void*)xmid, x, CC, CC);

  // 7) LN2 -> h2 (reuse hbuf)
  layernorm_kernel<false><<<dim3(CM / 8), 256, 0, stream>>>(xmid, gamma2, beta2, hbuf);

  // 8) FFN1 + GELU -> f
  gemm_bf16_kernel<EPI_GELU><<<dim3(CM / 64, CHID / 128), 256, 0, stream>>>(
      hbuf, w1T, b1, (void*)fbuf, nullptr, CHID, CC);

  // 9) depthwise 3x3 + GELU -> g
  dwconv_kernel<<<dim3(CM * (CHID / 8) / 256), 256, 0, stream>>>(fbuf, dw_w, dw_b, gbuf);

  // 10) FFN2 + residual -> d_out (fp32)
  gemm_bf16_kernel<EPI_RES><<<dim3(CM / 64, CC / 128), 256, 0, stream>>>(
      gbuf, w2T, b2, d_out, xmid, CC, CHID);
}
